// TransformerBlock_59176059404472
// MI455X (gfx1250) — matrix operand order repaired and checked by reading, not compiled, not run on hardware
//
#include <hip/hip_runtime.h>
#include <hip/hip_bf16.h>

// ---------------------------------------------------------------------------
// Types for CDNA5 WMMA (wave32): A/B = 16 bf16 per lane, C/D = 8 f32 per lane
// ---------------------------------------------------------------------------
typedef __attribute__((ext_vector_type(16))) __bf16 v16bf;
typedef __attribute__((ext_vector_type(8)))  float  v8f;
typedef __attribute__((ext_vector_type(4)))  int    v4i;

union FragU {
    v16bf v;
    unsigned short u[16];
    unsigned int   w[8];
};
union AccU {
    v8f   v;
    float f[8];
};

__device__ __forceinline__ unsigned short f2bf(float f) {
    unsigned int u = __float_as_uint(f);
    u += 0x7FFFu + ((u >> 16) & 1u);   // round-to-nearest-even
    return (unsigned short)(u >> 16);
}

// A-matrix (16-bit, 16x32) per-lane K mapping (ISA 7.12.2):
// lanes 0-15: VGPR0..3 -> K 0..7, VGPR4..7 -> K 16..23 ; lanes 16-31: +8
__device__ __forceinline__ int kpairA(int v, int hf) {
    return ((v >> 2) << 4) + (hf << 3) + ((v & 3) << 1);
}
// B-matrix (16-bit, 32x16): same K mapping as A (lanes 0-15: K 0..7,16..23; lanes 16-31: +8)
__device__ __forceinline__ int kpairB(int v, int hf) {
    return ((v >> 2) << 4) + (hf << 3) + ((v & 3) << 1);
}

__device__ __forceinline__ v8f zero8() {
    v8f z = {0.f, 0.f, 0.f, 0.f, 0.f, 0.f, 0.f, 0.f};
    return z;
}

// ---------------------------------------------------------------------------
// Optional CDNA5 async global->LDS path (ASYNCcnt). Compile-safe fallback.
// Builtin signature (from probe): (v4i addrspace(1)*, v4i addrspace(3)*, imm, imm)
// ---------------------------------------------------------------------------
#if defined(__HIP_DEVICE_COMPILE__) && \
    __has_builtin(__builtin_amdgcn_global_load_async_to_lds_b128) && \
    __has_builtin(__builtin_amdgcn_s_wait_asynccnt)
#define USE_ASYNC_LDS 1
typedef __attribute__((address_space(1))) v4i g_v4i;
typedef __attribute__((address_space(3))) v4i l_v4i;
#else
#define USE_ASYNC_LDS 0
#endif

// Copy 16 bytes (8 bf16) global -> LDS.
__device__ __forceinline__ void ldg_to_lds_b128(unsigned short* lds_dst,
                                                const unsigned short* g_src) {
#if USE_ASYNC_LDS
    __builtin_amdgcn_global_load_async_to_lds_b128(
        (g_v4i*)g_src, (l_v4i*)lds_dst, 0, 0);
#else
    *reinterpret_cast<uint4*>(lds_dst) = *reinterpret_cast<const uint4*>(g_src);
#endif
}
__device__ __forceinline__ void wait_async_lds() {
#if USE_ASYNC_LDS
    __builtin_amdgcn_s_wait_asynccnt(0);
#endif
}

// ---------------------------------------------------------------------------
// f32 -> bf16 conversion (weights / generic)
// ---------------------------------------------------------------------------
__global__ __launch_bounds__(256) void cvt_bf16_kernel(const float* __restrict__ in,
                                                       unsigned short* __restrict__ out,
                                                       int n) {
    for (int i = blockIdx.x * blockDim.x + threadIdx.x; i < n; i += gridDim.x * blockDim.x)
        out[i] = f2bf(in[i]);
}

// ---------------------------------------------------------------------------
// Row LayerNorm: f32 in -> bf16 out (feeds WMMA GEMMs)
// ---------------------------------------------------------------------------
__global__ __launch_bounds__(256) void layernorm_kernel(const float* __restrict__ x,
                                                        const float* __restrict__ g,
                                                        const float* __restrict__ b,
                                                        unsigned short* __restrict__ out,
                                                        int D) {
    const int row = blockIdx.x;
    const float* xr = x + (size_t)row * D;
    __shared__ float red[256];
    const int tid = threadIdx.x;

    float s = 0.f;
    for (int i = tid; i < D; i += 256) s += xr[i];
    red[tid] = s; __syncthreads();
    for (int off = 128; off > 0; off >>= 1) {
        if (tid < off) red[tid] += red[tid + off];
        __syncthreads();
    }
    const float mu = red[0] * (1.0f / D);
    __syncthreads();

    float s2 = 0.f;
    for (int i = tid; i < D; i += 256) { float d = xr[i] - mu; s2 += d * d; }
    red[tid] = s2; __syncthreads();
    for (int off = 128; off > 0; off >>= 1) {
        if (tid < off) red[tid] += red[tid + off];
        __syncthreads();
    }
    const float rstd = rsqrtf(red[0] * (1.0f / D) + 1e-5f);

    unsigned short* orow = out + (size_t)row * D;
    for (int i = tid; i < D; i += 256)
        orow[i] = f2bf((xr[i] - mu) * rstd * g[i] + b[i]);
}

// ---------------------------------------------------------------------------
// WMMA GEMM: C[M,N] = A[M,K](bf16) * B[K,N](bf16) (+bias)(+resid)(+gelu)
// Block tile 128x128 (8 waves of 32x64 = 2x4 WMMA tiles), K-step 32.
// LDS: A row-major (async b128 when available), B transposed for contiguous
// ds_load_b32 fragment reads. 8 WMMAs per wave per K-step.
// EPI 0: bf16 out.  EPI 1: +bias, exact GELU, bf16 out.  EPI 2: (+bias)+resid, f32 out.
// ---------------------------------------------------------------------------
template <int EPI>
__global__ __launch_bounds__(256) void wmma_gemm_kernel(
    const unsigned short* __restrict__ A,
    const unsigned short* __restrict__ Bm,
    const float* __restrict__ bias,
    const float* __restrict__ resid,
    float* __restrict__ Cf,
    unsigned short* __restrict__ Cb,
    int M, int N, int K)
{
    __shared__ __align__(16) unsigned short As[128 * 32];
    __shared__ __align__(16) unsigned short Bs[128 * 32];   // transposed: Bs[n][k]

    const int tid  = threadIdx.x;
    const int lane = tid & 31;
    const int wid  = tid >> 5;
    const int m    = lane & 15;
    const int hf   = lane >> 4;
    const int wm   = wid & 3;      // 0..3  -> row offset wm*32
    const int wn   = wid >> 2;     // 0..1  -> col offset wn*64
    const int m0g  = blockIdx.y * 128;
    const int n0g  = blockIdx.x * 128;

    AccU acc[2][4];
    for (int i = 0; i < 2; ++i)
        for (int j = 0; j < 4; ++j)
            acc[i][j].v = zero8();

    const int nB  = tid & 127;     // B staging: col
    const int kgB = tid >> 7;      // 0..1

    for (int kb = 0; kb < K; kb += 32) {
        // A tile 128x32: 512 x b128, 2 per thread (async -> LDS when available)
        for (int i = 0; i < 2; ++i) {
            const int li  = tid + i * 256;
            const int row = li >> 2;
            const int kq  = (li & 3) << 3;
            ldg_to_lds_b128(&As[li << 3],
                            A + (size_t)(m0g + row) * K + kb + kq);
        }
        // B tile 32x128 transposed into LDS (global reads coalesced over n)
        for (int i = 0; i < 16; ++i) {
            const int k = kgB * 16 + i;
            Bs[nB * 32 + k] = Bm[(size_t)(kb + k) * N + n0g + nB];
        }
        wait_async_lds();
        __syncthreads();

        FragU aF[2], bF[4];
        for (int i = 0; i < 2; ++i) {
            const int arow = wm * 32 + i * 16 + m;
            for (int v = 0; v < 8; ++v)
                aF[i].w[v] = *reinterpret_cast<const unsigned int*>(&As[arow * 32 + kpairA(v, hf)]);
        }
        for (int j = 0; j < 4; ++j) {
            const int bcol = wn * 64 + j * 16 + m;
            for (int v = 0; v < 8; ++v)
                bF[j].w[v] = *reinterpret_cast<const unsigned int*>(&Bs[bcol * 32 + kpairB(v, hf)]);
        }
        for (int i = 0; i < 2; ++i)
            for (int j = 0; j < 4; ++j)
                acc[i][j].v = __builtin_amdgcn_wmma_f32_16x16x32_bf16(
                    false, aF[i].v, false, bF[j].v, (short)0, acc[i][j].v, false, false);
        __syncthreads();
    }

    // Epilogue (C layout: VGPR r -> row r (lanes 0-15) / r+8 (lanes 16-31), col = lane%16)
    for (int i = 0; i < 2; ++i)
        for (int j = 0; j < 4; ++j)
            for (int r = 0; r < 8; ++r) {
                const int row = m0g + wm * 32 + i * 16 + r + (hf << 3);
                const int col = n0g + wn * 64 + j * 16 + m;
                float vv = acc[i][j].f[r];
                if (EPI == 1) {
                    vv += bias[col];
                    vv = 0.5f * vv * (1.0f + erff(vv * 0.70710678118654752f));
                }
                if (EPI == 2) {
                    if (bias) vv += bias[col];
                    vv += resid[(size_t)row * N + col];
                    Cf[(size_t)row * N + col] = vv;
                } else {
                    Cb[(size_t)row * N + col] = f2bf(vv);
                }
            }
}

// ---------------------------------------------------------------------------
// Causal flash attention. 4 waves per block share one (batch, head): each wave
// owns a 16-query tile; the block cooperatively stages each 32-key chunk of
// K (row-major, b128/async) and V (transposed Vt[dk][key]) in LDS, so every
// fragment read is 8 contiguous ds_load_b32. Online softmax uses 16-lane
// shfl reductions; P goes C-layout -> A-layout via a per-wave LDS bounce.
// Causal guards are wave-uniform, so EXEC stays all-1s around every WMMA.
// ---------------------------------------------------------------------------
__global__ __launch_bounds__(128) void attn_kernel(
    const unsigned short* __restrict__ Qb,
    const unsigned short* __restrict__ Kb,
    const unsigned short* __restrict__ Vb,
    unsigned short* __restrict__ Ob,
    int S, int H, int D)
{
    __shared__ __align__(16) unsigned short Ks[32 * 64];   // [key][dk]
    __shared__ __align__(16) unsigned short Vt[64 * 32];   // [dk][key]
    __shared__ __align__(16) unsigned short Pball[4][16 * 32];

    const int nqg = S >> 6;                    // groups of 4 query tiles
    const int grp = blockIdx.x % nqg;
    const int h   = (blockIdx.x / nqg) % H;
    const int b   = blockIdx.x / (nqg * H);

    const int tid  = threadIdx.x;
    const int lane = tid & 31;
    const int wid  = tid >> 5;                 // 0..3: query tile within group
    const int m    = lane & 15;
    const int hf   = lane >> 4;
    const int hf8  = hf << 3;
    const int q0   = ((grp << 2) + wid) << 4;  // this wave's query-tile base
    const int hc   = h * 64;
    unsigned short* Pb = Pball[wid];

    // Q fragments (A layout): rows = queries, K dim = Dk (two steps of 32)
    FragU qf[2];
    for (int ks = 0; ks < 2; ++ks)
        for (int v = 0; v < 8; ++v)
            qf[ks].w[v] = *reinterpret_cast<const unsigned int*>(
                Qb + (size_t)(b * S + q0 + m) * D + hc + ks * 32 + kpairA(v, hf));

    float mrow[8], lrow[8];
    AccU Oa[4];
    for (int r = 0; r < 8; ++r) { mrow[r] = -1e30f; lrow[r] = 0.f; }
    for (int t = 0; t < 4; ++t) Oa[t].v = zero8();

    const float scale = 0.125f;                      // 1/sqrt(64)
    const int my_nch  = (q0 + 47) >> 5;              // chunks this wave needs
    const int nch_max = ((((grp << 2) + 3) << 4) + 47) >> 5;

    for (int ch = 0; ch < nch_max; ++ch) {
        __syncthreads();   // protect LDS tiles from previous iteration's readers

        // Stage K chunk 32x64: 256 x b128, 2 per thread
        {
            const size_t krow0 = (size_t)(b * S + (ch << 5));
            for (int i = 0; i < 2; ++i) {
                const int li  = tid + i * 128;       // 0..255
                const int key = li >> 3;
                const int dkq = (li & 7) << 3;
                ldg_to_lds_b128(&Ks[li << 3],
                                Kb + (krow0 + key) * D + hc + dkq);
            }
        }
        // Stage V chunk transposed: thread handles one key x 16 dk
        {
            const int keyv = tid & 31;
            const int dk0  = (tid >> 5) << 4;        // 0,16,32,48
            const unsigned short* vp =
                Vb + (size_t)(b * S + (ch << 5) + keyv) * D + hc + dk0;
            union { uint4 q[2]; unsigned short s[16]; } uv;
            uv.q[0] = *reinterpret_cast<const uint4*>(vp);
            uv.q[1] = *reinterpret_cast<const uint4*>(vp + 8);
            for (int e = 0; e < 16; ++e)
                Vt[(dk0 + e) * 32 + keyv] = uv.s[e];
        }
        wait_async_lds();
        __syncthreads();

        if (ch < my_nch) {   // wave-uniform: EXEC stays all-1s inside
            // K^T fragments (B layout): n = key position, K dim = Dk
            FragU kf[2][2];
            for (int t = 0; t < 2; ++t)
                for (int ks = 0; ks < 2; ++ks)
                    for (int v = 0; v < 8; ++v)
                        kf[t][ks].w[v] = *reinterpret_cast<const unsigned int*>(
                            &Ks[(t * 16 + m) * 64 + ks * 32 + kpairB(v, hf)]);

            AccU sa[2];
            sa[0].v = zero8(); sa[1].v = zero8();
            for (int ks = 0; ks < 2; ++ks) {
                sa[0].v = __builtin_amdgcn_wmma_f32_16x16x32_bf16(
                    false, qf[ks].v, false, kf[0][ks].v, (short)0, sa[0].v, false, false);
                sa[1].v = __builtin_amdgcn_wmma_f32_16x16x32_bf16(
                    false, qf[ks].v, false, kf[1][ks].v, (short)0, sa[1].v, false, false);
            }

            // Online softmax per row (rows live in 16-lane groups per VGPR)
            for (int r = 0; r < 8; ++r) {
                const int qrow = q0 + r + hf8;
                float s0 = sa[0].f[r] * scale; if ((ch << 5) + m      > qrow) s0 = -1e30f;
                float s1 = sa[1].f[r] * scale; if ((ch << 5) + 16 + m > qrow) s1 = -1e30f;

                float vmx = fmaxf(s0, s1);
                vmx = fmaxf(vmx, __shfl_xor(vmx, 1, 32));
                vmx = fmaxf(vmx, __shfl_xor(vmx, 2, 32));
                vmx = fmaxf(vmx, __shfl_xor(vmx, 4, 32));
                vmx = fmaxf(vmx, __shfl_xor(vmx, 8, 32));

                const float mn = fmaxf(mrow[r], vmx);
                const float al = __expf(mrow[r] - mn);
                const float p0 = __expf(s0 - mn);
                const float p1 = __expf(s1 - mn);
                float ps = p0 + p1;
                ps += __shfl_xor(ps, 1, 32);
                ps += __shfl_xor(ps, 2, 32);
                ps += __shfl_xor(ps, 4, 32);
                ps += __shfl_xor(ps, 8, 32);
                lrow[r] = lrow[r] * al + ps;
                mrow[r] = mn;
                for (int t = 0; t < 4; ++t) Oa[t].f[r] *= al;

                Pb[(r + hf8) * 32 + m]      = f2bf(p0);
                Pb[(r + hf8) * 32 + 16 + m] = f2bf(p1);
            }

            // P as A-fragment (LDS in-order within a wave)
            FragU pf;
            for (int v = 0; v < 8; ++v)
                pf.w[v] = *reinterpret_cast<const unsigned int*>(&Pb[m * 32 + kpairA(v, hf)]);

            // V fragments from transposed LDS: contiguous pairs
            for (int t = 0; t < 4; ++t) {
                FragU vf;
                for (int v = 0; v < 8; ++v)
                    vf.w[v] = *reinterpret_cast<const unsigned int*>(
                        &Vt[(t * 16 + m) * 32 + kpairB(v, hf)]);
                Oa[t].v = __builtin_amdgcn_wmma_f32_16x16x32_bf16(
                    false, pf.v, false, vf.v, (short)0, Oa[t].v, false, false);
            }
        }
    }

    for (int t = 0; t < 4; ++t)
        for (int r = 0; r < 8; ++r) {
            const float vv = Oa[t].f[r] / lrow[r];
            const size_t row = (size_t)(b * S + q0 + r + hf8);
            Ob[row * D + hc + t * 16 + m] = f2bf(vv);
        }
}

// ---------------------------------------------------------------------------
// Host-side pipeline
// ---------------------------------------------------------------------------
extern "C" void kernel_launch(void* const* d_in, const int* in_sizes, int n_in,
                              void* d_out, int out_size, void* d_ws, size_t ws_size,
                              hipStream_t stream) {
    (void)in_sizes; (void)n_in; (void)out_size; (void)ws_size;
    const float* x   = (const float*)d_in[0];
    const float* Wq  = (const float*)d_in[1];
    const float* Wk  = (const float*)d_in[2];
    const float* Wv  = (const float*)d_in[3];
    const float* Wo  = (const float*)d_in[4];
    const float* g1  = (const float*)d_in[5];
    const float* be1 = (const float*)d_in[6];
    const float* g2  = (const float*)d_in[7];
    const float* be2 = (const float*)d_in[8];
    const float* W1  = (const float*)d_in[9];
    const float* bf1 = (const float*)d_in[10];
    const float* W2  = (const float*)d_in[11];
    const float* bf2 = (const float*)d_in[12];
    float* out = (float*)d_out;

    constexpr int Bb = 2, S = 2048, D = 1024, H = 16, DFF = 4096;
    constexpr int BS = Bb * S;   // 4096 rows

    size_t off = 0;
    auto alloc = [&](size_t bytes) -> void* {
        void* p = (char*)d_ws + off;
        off += (bytes + 255) & ~(size_t)255;
        return p;
    };
    unsigned short* Wqb  = (unsigned short*)alloc((size_t)D * D * 2);
    unsigned short* Wkb  = (unsigned short*)alloc((size_t)D * D * 2);
    unsigned short* Wvb  = (unsigned short*)alloc((size_t)D * D * 2);
    unsigned short* Wob  = (unsigned short*)alloc((size_t)D * D * 2);
    unsigned short* W1b  = (unsigned short*)alloc((size_t)D * DFF * 2);
    unsigned short* W2b  = (unsigned short*)alloc((size_t)DFF * D * 2);
    unsigned short* ln1o = (unsigned short*)alloc((size_t)BS * D * 2);
    unsigned short* Qb   = (unsigned short*)alloc((size_t)BS * D * 2);
    unsigned short* Kbuf = (unsigned short*)alloc((size_t)BS * D * 2);
    unsigned short* Vbuf = (unsigned short*)alloc((size_t)BS * D * 2);
    unsigned short* Ab   = (unsigned short*)alloc((size_t)BS * D * 2);
    float*          x1   = (float*)        alloc((size_t)BS * D * 4);
    unsigned short* ln2o = (unsigned short*)alloc((size_t)BS * D * 2);
    unsigned short* h1   = (unsigned short*)alloc((size_t)BS * DFF * 2);

    // Convert weights to bf16
    cvt_bf16_kernel<<<1024, 256, 0, stream>>>(Wq, Wqb, D * D);
    cvt_bf16_kernel<<<1024, 256, 0, stream>>>(Wk, Wkb, D * D);
    cvt_bf16_kernel<<<1024, 256, 0, stream>>>(Wv, Wvb, D * D);
    cvt_bf16_kernel<<<1024, 256, 0, stream>>>(Wo, Wob, D * D);
    cvt_bf16_kernel<<<2048, 256, 0, stream>>>(W1, W1b, D * DFF);
    cvt_bf16_kernel<<<2048, 256, 0, stream>>>(W2, W2b, DFF * D);

    // LN1
    layernorm_kernel<<<BS, 256, 0, stream>>>(x, g1, be1, ln1o, D);

    // Q/K/V projections
    dim3 gD(D / 128, BS / 128);         // (8, 32)
    wmma_gemm_kernel<0><<<gD, 256, 0, stream>>>(ln1o, Wqb, nullptr, nullptr, nullptr, Qb,   BS, D, D);
    wmma_gemm_kernel<0><<<gD, 256, 0, stream>>>(ln1o, Wkb, nullptr, nullptr, nullptr, Kbuf, BS, D, D);
    wmma_gemm_kernel<0><<<gD, 256, 0, stream>>>(ln1o, Wvb, nullptr, nullptr, nullptr, Vbuf, BS, D, D);

    // Causal attention: 4 query tiles per block sharing K/V staging
    attn_kernel<<<Bb * H * (S / 64), 128, 0, stream>>>(Qb, Kbuf, Vbuf, Ab, S, H, D);

    // Output projection + residual -> x1 (f32)
    wmma_gemm_kernel<2><<<gD, 256, 0, stream>>>(Ab, Wob, nullptr, x, x1, nullptr, BS, D, D);

    // LN2
    layernorm_kernel<<<BS, 256, 0, stream>>>(x1, g2, be2, ln2o, D);

    // FFN: GELU(h @ W1 + b1) @ W2 + b2 + x1 -> out (f32)
    dim3 gFF(DFF / 128, BS / 128);      // (32, 32)
    wmma_gemm_kernel<1><<<gFF, 256, 0, stream>>>(ln2o, W1b, bf1, nullptr, nullptr, h1, BS, DFF, D);
    wmma_gemm_kernel<2><<<gD,  256, 0, stream>>>(h1,   W2b, bf2, x1,      out,     nullptr, BS, D, DFF);
}
